// GNNV2Anomaly_841813590020
// MI455X (gfx1250) — matrix operand order repaired and checked by reading, not compiled, not run on hardware
//
#include <hip/hip_runtime.h>
#include <hip/hip_bf16.h>
#include <math.h>

// Problem constants (from reference)
#define NN 50000
#define EE 800000
#define DD 256
#define HH 256
#define LL 64
#define ENE (EE + NN)   // edges + self loops
#define NEG_SLOPE 0.2f

typedef unsigned int u32;
typedef unsigned short u16;
typedef __attribute__((ext_vector_type(16))) __bf16 v16bf;
typedef __attribute__((ext_vector_type(8)))  float  v8f;

union BFrag { uint4 q[2]; v16bf v; };

__device__ __forceinline__ u16 f2bf(float f) {
  u32 u = __float_as_uint(f);
  u32 r = u + 0x7FFFu + ((u >> 16) & 1u);   // round-to-nearest-even
  return (u16)(r >> 16);
}
// monotonic float <-> uint key for atomicMax-based segment max
__device__ __forceinline__ u32 fkey(float f) {
  u32 b = __float_as_uint(f);
  return (b & 0x80000000u) ? ~b : (b ^ 0x80000000u);
}
__device__ __forceinline__ float unkey(u32 k) {
  u32 b = (k & 0x80000000u) ? (k ^ 0x80000000u) : ~k;
  return __uint_as_float(b);
}
__device__ __forceinline__ float lrelu(float x) { return x > 0.f ? x : NEG_SLOPE * x; }
__device__ __forceinline__ float gelu_exact(float x) {
  return 0.5f * x * (1.f + erff(x * 0.70710678118654752f));
}

// ---------------- elementwise kernels ----------------
__global__ void k_fill0(float* p, long n) {
  long i = (long)blockIdx.x * 256 + threadIdx.x;
  long st = (long)gridDim.x * 256;
  for (; i < n; i += st) p[i] = 0.f;
}

__global__ void k_cvt_bf16(const float* __restrict__ in, u16* __restrict__ out, long n) {
  long i = (long)blockIdx.x * 256 + threadIdx.x;
  if (i < n) out[i] = f2bf(in[i]);
}

// Pack fp32 row-major W[K,Nc] into per-(ktile,ntile) WMMA B-fragment order:
// dword index = ((kt*nTiles + nt)*32 + lane)*8 + v
// lane: col = nt*16 + (lane&15); khalf = (lane>>4)*16; VGPR v holds (k0, k0+1)
__global__ void k_pack_w(const float* __restrict__ W, u32* __restrict__ P, int K, int Nc) {
  int total = (K / 32) * (Nc / 16) * 256;
  int i = blockIdx.x * 256 + threadIdx.x;
  if (i >= total) return;
  int nT = Nc / 16;
  int kt = i / (nT * 256);
  int rem = i - kt * nT * 256;
  int nt = rem >> 8;
  int r2 = rem & 255;
  int lane = r2 >> 3, v = r2 & 7;
  int c = nt * 16 + (lane & 15);
  int k0 = kt * 32 + ((lane >> 4) << 4) + 2 * v;
  u32 lo = f2bf(W[(size_t)k0 * Nc + c]);
  u32 hi = f2bf(W[(size_t)(k0 + 1) * Nc + c]);
  P[i] = lo | (hi << 16);
}

// ---------------- WMMA GEMM: C[M,Nc] = A_bf16[M,K] @ Wpacked + bias ----------------
// 8 waves/block, each wave owns one 16-row tile of the same 16-col tile.
// B strip for this column tile staged in LDS once, shared by all waves.
template <int K>
__global__ __launch_bounds__(256) void k_gemm_wmma(const u16* __restrict__ A,
                                                   const u32* __restrict__ Wp,
                                                   const float* __restrict__ bias,
                                                   float* __restrict__ C,
                                                   int M, int Nc) {
  constexpr int NKT = K / 32;
  __shared__ u32 ldsW[NKT * 256];
  const int nT = Nc >> 4;
  const int nt = blockIdx.y;
  const int tid = threadIdx.x;

  const uint4* Wp4 = (const uint4*)Wp;
  uint4* lds4 = (uint4*)ldsW;
  for (int i = tid; i < NKT * 64; i += 256) {
    int kt = i >> 6;
    lds4[i] = Wp4[((size_t)(kt * nT + nt) << 6) + (i & 63)];
  }
  __syncthreads();

  const int wave = tid >> 5, lane = tid & 31;
  const int rowTile = blockIdx.x * 8 + wave;
  if (rowTile * 16 >= M) return;
  const int laneHi = lane >> 4;
  const int row = rowTile * 16 + (lane & 15);
  const char* Arow = (const char*)(A + (size_t)row * K);

  v8f acc = {};
#pragma unroll
  for (int kt = 0; kt < NKT; ++kt) {
    BFrag a, b;
    // A 16x32, contiguous k order (same as packed B): lanes 0-15 K={0..15}, lanes 16-31 K={16..31}
    const char* ap = Arow + ((kt << 5) + (laneHi << 4)) * 2;
    a.q[0] = *(const uint4*)ap;
    a.q[1] = *(const uint4*)(ap + 16);
    b.q[0] = lds4[(kt << 6) + lane * 2];
    b.q[1] = lds4[(kt << 6) + lane * 2 + 1];
    if (kt + 1 < NKT) __builtin_prefetch(ap + 64, 0, 3);  // next K-chunk of this row
    acc = __builtin_amdgcn_wmma_f32_16x16x32_bf16(false, a.v, false, b.v,
                                                  (short)0, acc, false, false);
  }

  // C/D layout: VGPR r, lane -> M = r + 8*(lane>=16), N = lane&15
  const int col = nt * 16 + (lane & 15);
  const float bv = bias ? bias[col] : 0.f;
  const int rbase = rowTile * 16 + laneHi * 8;
#pragma unroll
  for (int r = 0; r < 8; ++r)
    C[(size_t)(rbase + r) * Nc + col] = acc[r] + bv;
}

// ---------------- GATv2 edge kernels (wave32: one wave per edge) ----------------
__global__ __launch_bounds__(256) void k_edge_scores(const float* __restrict__ xl,
                                                     const float* __restrict__ xr,
                                                     const int* __restrict__ srcA,
                                                     const int* __restrict__ dstA,
                                                     const float* __restrict__ att,
                                                     float* __restrict__ ebuf,
                                                     u32* __restrict__ mkey) {
  const int lane = threadIdx.x & 31;
  const long eid = (long)blockIdx.x * 8 + (threadIdx.x >> 5);
  if (eid >= ENE) return;
  const int src = (eid < EE) ? srcA[eid] : (int)(eid - EE);
  const int dst = (eid < EE) ? dstA[eid] : (int)(eid - EE);
  const float4* a = (const float4*)(xl + (size_t)src * HH) + lane * 2;
  const float4* b = (const float4*)(xr + (size_t)dst * HH) + lane * 2;
  const float4* t = (const float4*)att + lane * 2;
  float v = 0.f;
#pragma unroll
  for (int j = 0; j < 2; ++j) {
    float4 A = a[j], B = b[j], T = t[j];
    v += lrelu(A.x + B.x) * T.x + lrelu(A.y + B.y) * T.y +
         lrelu(A.z + B.z) * T.z + lrelu(A.w + B.w) * T.w;
  }
  for (int off = 16; off > 0; off >>= 1) v += __shfl_xor(v, off, 32);
  if (lane == 0) {
    ebuf[eid] = v;
    atomicMax(&mkey[dst], fkey(v));   // segment max, served out of L2
  }
}

__global__ void k_edge_ps(const float* __restrict__ ebuf, float* __restrict__ pbuf,
                          const u32* __restrict__ mkey, float* __restrict__ sden,
                          const int* __restrict__ dstA) {
  long eid = (long)blockIdx.x * 256 + threadIdx.x;
  if (eid >= ENE) return;
  int dst = (eid < EE) ? dstA[eid] : (int)(eid - EE);
  float p = expf(ebuf[eid] - unkey(mkey[dst]));
  pbuf[eid] = p;
  atomicAdd(&sden[dst], p);
}

__global__ __launch_bounds__(256) void k_edge_agg(const float* __restrict__ xl,
                                                  const float* __restrict__ pbuf,
                                                  const float* __restrict__ sden,
                                                  const int* __restrict__ srcA,
                                                  const int* __restrict__ dstA,
                                                  float* __restrict__ out) {
  const int lane = threadIdx.x & 31;
  const long eid = (long)blockIdx.x * 8 + (threadIdx.x >> 5);
  if (eid >= ENE) return;
  const int src = (eid < EE) ? srcA[eid] : (int)(eid - EE);
  const int dst = (eid < EE) ? dstA[eid] : (int)(eid - EE);
  const float alpha = pbuf[eid] / sden[dst];
  const float4* h = (const float4*)(xl + (size_t)src * HH) + lane * 2;
  float* o = out + (size_t)dst * HH + lane * 8;
  float4 h0 = h[0], h1 = h[1];
  atomicAdd(o + 0, alpha * h0.x); atomicAdd(o + 1, alpha * h0.y);
  atomicAdd(o + 2, alpha * h0.z); atomicAdd(o + 3, alpha * h0.w);
  atomicAdd(o + 4, alpha * h1.x); atomicAdd(o + 5, alpha * h1.y);
  atomicAdd(o + 6, alpha * h1.z); atomicAdd(o + 7, alpha * h1.w);
}

__global__ void k_bias_gelu(float* __restrict__ h, const float* __restrict__ b, long n) {
  long i = (long)blockIdx.x * 256 + threadIdx.x;
  if (i >= n) return;
  h[i] = gelu_exact(h[i] + b[i & (HH - 1)]);
}

__global__ void k_combine(const float* __restrict__ eps, float* __restrict__ muO,
                          float* __restrict__ lsO, float* __restrict__ zO,
                          float* __restrict__ zf) {
  long i = (long)blockIdx.x * 256 + threadIdx.x;
  if (i >= (long)NN * LL) return;
  float mu = muO[i];
  float ls = fminf(lsO[i], 10.f);
  lsO[i] = ls;
  float z = mu + eps[i] * expf(ls);
  zO[i] = z;
  zf[i] = z;
}

// ---------------- host driver ----------------
extern "C" void kernel_launch(void* const* d_in, const int* in_sizes, int n_in,
                              void* d_out, int out_size, void* d_ws, size_t ws_size,
                              hipStream_t stream) {
  (void)in_sizes; (void)n_in; (void)out_size; (void)ws_size;
  const float* x        = (const float*)d_in[0];
  const int*   ei       = (const int*)  d_in[1];
  const float* eps      = (const float*)d_in[2];
  const float* enc_Wl   = (const float*)d_in[3];
  const float* enc_Wr   = (const float*)d_in[4];
  const float* enc_att  = (const float*)d_in[5];
  const float* enc_b    = (const float*)d_in[6];
  const float* mu_W     = (const float*)d_in[7];
  const float* mu_b     = (const float*)d_in[8];
  const float* ls_W     = (const float*)d_in[9];
  const float* ls_b     = (const float*)d_in[10];
  const float* dec0_Wl  = (const float*)d_in[11];
  const float* dec0_Wr  = (const float*)d_in[12];
  const float* dec0_att = (const float*)d_in[13];
  const float* dec0_b   = (const float*)d_in[14];
  const float* dec_Wl   = (const float*)d_in[15];
  const float* dec_Wr   = (const float*)d_in[16];
  const float* dec_att  = (const float*)d_in[17];
  const float* dec_b    = (const float*)d_in[18];
  const float* out_W    = (const float*)d_in[19];
  const float* out_b    = (const float*)d_in[20];

  const int* srcA = ei;
  const int* dstA = ei + EE;

  // workspace carve
  char* wp = (char*)d_ws;
  auto carve = [&](size_t bytes) {
    void* p = (void*)wp;
    wp += (bytes + 255) & ~(size_t)255;
    return p;
  };
  u16*   bfA  = (u16*)  carve((size_t)NN * HH * sizeof(u16));   // bf16 activations
  float* xl   = (float*)carve((size_t)NN * HH * sizeof(float));
  float* xr   = (float*)carve((size_t)NN * HH * sizeof(float));
  float* hb   = (float*)carve((size_t)NN * HH * sizeof(float)); // aggregation out
  float* zf   = (float*)carve((size_t)NN * LL * sizeof(float));
  float* ebuf = (float*)carve((size_t)ENE * sizeof(float));
  float* pbuf = (float*)carve((size_t)ENE * sizeof(float));
  u32*   mkey = (u32*)  carve((size_t)NN * sizeof(u32));
  float* sden = (float*)carve((size_t)NN * sizeof(float));
  u32* packBig = (u32*)carve((size_t)15 * 32768 * sizeof(u32)); // 15 x (256x256)
  u32* packSm  = (u32*)carve((size_t)4 * 8192 * sizeof(u32));   // mu, ls, dec0 x2

  u32* encWlP[4]; u32* encWrP[4]; u32* decWlP[3]; u32* decWrP[3];
  for (int l = 0; l < 4; ++l) { encWlP[l] = packBig + (size_t)l * 32768;
                                encWrP[l] = packBig + (size_t)(4 + l) * 32768; }
  for (int l = 0; l < 3; ++l) { decWlP[l] = packBig + (size_t)(8 + l) * 32768;
                                decWrP[l] = packBig + (size_t)(11 + l) * 32768; }
  u32* outP = packBig + (size_t)14 * 32768;
  u32* muP  = packSm;
  u32* lsP  = packSm + 8192;
  u32* d0lP = packSm + 2 * 8192;
  u32* d0rP = packSm + 3 * 8192;

  auto packW = [&](const float* W, u32* P, int K, int Nc) {
    int total = (K / 32) * (Nc / 16) * 256;
    k_pack_w<<<(total + 255) / 256, 256, 0, stream>>>(W, P, K, Nc);
  };
  auto fill0 = [&](float* p, long n) { k_fill0<<<2048, 256, 0, stream>>>(p, n); };
  auto cvt = [&](const float* in, long n) {
    k_cvt_bf16<<<(int)((n + 255) / 256), 256, 0, stream>>>(in, bfA, n);
  };
  auto gemm256 = [&](const u32* P, const float* bias, float* C, int Nc) {
    dim3 g((3125 + 7) / 8, Nc / 16);
    k_gemm_wmma<256><<<g, 256, 0, stream>>>(bfA, P, bias, C, NN, Nc);
  };
  auto gemm64 = [&](const u32* P, const float* bias, float* C, int Nc) {
    dim3 g((3125 + 7) / 8, Nc / 16);
    k_gemm_wmma<64><<<g, 256, 0, stream>>>(bfA, P, bias, C, NN, Nc);
  };

  auto gat = [&](int K, const u32* WlP, const u32* WrP, const float* att, const float* b) {
    if (K == 256) { gemm256(WlP, nullptr, xl, HH); gemm256(WrP, nullptr, xr, HH); }
    else          { gemm64 (WlP, nullptr, xl, HH); gemm64 (WrP, nullptr, xr, HH); }
    fill0(hb, (long)NN * HH);
    fill0(sden, NN);
    fill0((float*)mkey, NN);        // key 0 == lowest possible key
    k_edge_scores<<<(ENE + 7) / 8, 256, 0, stream>>>(xl, xr, srcA, dstA, att, ebuf, mkey);
    k_edge_ps<<<(ENE + 255) / 256, 256, 0, stream>>>(ebuf, pbuf, mkey, sden, dstA);
    k_edge_agg<<<(ENE + 7) / 8, 256, 0, stream>>>(xl, pbuf, sden, srcA, dstA, hb);
    k_bias_gelu<<<(int)(((long)NN * HH + 255) / 256), 256, 0, stream>>>(hb, b, (long)NN * HH);
    cvt(hb, (long)NN * HH);         // next layer's bf16 input
  };

  // ---- pack all weights (cheap, deterministic every call) ----
  for (int l = 0; l < 4; ++l) {
    packW(enc_Wl + (size_t)l * DD * HH, encWlP[l], 256, 256);
    packW(enc_Wr + (size_t)l * DD * HH, encWrP[l], 256, 256);
  }
  packW(mu_W, muP, 256, 64);
  packW(ls_W, lsP, 256, 64);
  packW(dec0_Wl, d0lP, 64, 256);
  packW(dec0_Wr, d0rP, 64, 256);
  for (int l = 0; l < 3; ++l) {
    packW(dec_Wl + (size_t)l * HH * HH, decWlP[l], 256, 256);
    packW(dec_Wr + (size_t)l * HH * HH, decWrP[l], 256, 256);
  }
  packW(out_W, outP, 256, 256);

  // ---- encoder ----
  cvt(x, (long)NN * DD);
  for (int l = 0; l < 4; ++l)
    gat(256, encWlP[l], encWrP[l], enc_att + (size_t)l * HH, enc_b + (size_t)l * HH);

  // ---- latent head: mu / logstd / z straight into d_out slices ----
  float* outF = (float*)d_out;
  float* zO  = outF + (size_t)NN * DD;
  float* muO = zO + (size_t)NN * LL;
  float* lsO = muO + (size_t)NN * LL;
  gemm256(muP, mu_b, muO, LL);
  gemm256(lsP, ls_b, lsO, LL);
  k_combine<<<(int)(((long)NN * LL + 255) / 256), 256, 0, stream>>>(eps, muO, lsO, zO, zf);

  // ---- decoder ----
  cvt(zf, (long)NN * LL);
  gat(64, d0lP, d0rP, dec0_att, dec0_b);
  for (int l = 0; l < 3; ++l)
    gat(256, decWlP[l], decWrP[l], dec_att + (size_t)l * HH, dec_b + (size_t)l * HH);

  // ---- reconstruction: x_rec = h @ out_W + out_b ----
  gemm256(outP, out_b, outF, DD);
}
